// GRUVAE_64089501991276
// MI455X (gfx1250) — matrix operand order repaired and checked by reading, not compiled, not run on hardware
//
#include <hip/hip_runtime.h>
#include <hip/hip_bf16.h>

typedef __attribute__((ext_vector_type(16))) _Float16 v16h;
typedef __attribute__((ext_vector_type(8)))  _Float16 v8h;
typedef __attribute__((ext_vector_type(8)))  float    v8f;
typedef __attribute__((ext_vector_type(4)))  unsigned int v4u;
typedef __attribute__((ext_vector_type(8)))  int v8i;
typedef __attribute__((ext_vector_type(4)))  int v4i;

__device__ __forceinline__ float wave_reduce(float v) {
#pragma unroll
  for (int off = 16; off > 0; off >>= 1) v += __shfl_down(v, off, 32);
  return __shfl(v, 0, 32);
}

__device__ __forceinline__ v16h cat8(v8h lo, v8h hi) {
  return __builtin_shufflevector(lo, hi, 0,1,2,3,4,5,6,7,8,9,10,11,12,13,14,15);
}

// A-matrix (16xK, lane = M row) fragment for one K=32 slab.
// Layout (ISA 7.12.2): lanes 0-15 hold K[0..7] & K[16..23]; lanes 16-31 K[8..15] & K[24..31].
__device__ __forceinline__ v16h load_a_frag(const _Float16* rowk, int khalf /*0|8*/) {
  v8h lo = *(const v8h*)(rowk + khalf);
  v8h hi = *(const v8h*)(rowk + 16 + khalf);
  return cat8(lo, hi);
}

// B-matrix (Kx16, lane = N col) fragment: lanes 0-15 hold K[0..7] & K[16..23], lanes 16-31 K[8..15] & K[24..31].
__device__ __forceinline__ v16h load_b_frag(const _Float16* colk, int kb /*0|16*/) {
  v8h lo = *(const v8h*)(colk + (kb >> 1));
  v8h hi = *(const v8h*)(colk + 16 + (kb >> 1));
  return cat8(lo, hi);
}

// ---------------------------------------------------------------------------
// Tensor Data Mover: DMA a 2D tile (tile_d1 rows x tile_d0 elements, f16) from
// global memory into LDS.  D# packed per CDNA5 ISA ch.8 (group0: count/lds/
// global addr/type=2; group1: data_size=2B, dims, tile dims, strides).
// Tracked with TENSORcnt.
// ---------------------------------------------------------------------------
__device__ __forceinline__ void tdm_load_2d_f16(unsigned lds_addr, const _Float16* gptr,
                                                unsigned tensor_d0, unsigned tensor_d1,
                                                unsigned tile_d0, unsigned tile_d1,
                                                unsigned long long stride0)
{
  unsigned long long ga = (unsigned long long)(const void*)gptr;
  v4u g0;
  g0.x = 1u;                                            // count=1, is_restore=0
  g0.y = lds_addr;                                      // bits[63:32]  lds_addr
  g0.z = (unsigned)(ga & 0xFFFFFFFFu);                  // bits[95:64]  global_addr lo
  g0.w = (unsigned)((ga >> 32) & 0x1FFFFFFu) | (2u << 30); // global_addr hi + type=2
  v8i g1;
  g1[0] = (int)(1u << 16);                              // data_size=1 (2 bytes/elem)
  g1[1] = (int)((tensor_d0 & 0xFFFFu) << 16);           // tensor_dim0[15:0]
  g1[2] = (int)(((tensor_d0 >> 16) & 0xFFFFu) | ((tensor_d1 & 0xFFFFu) << 16));
  g1[3] = (int)(((tensor_d1 >> 16) & 0xFFFFu) | ((tile_d0 & 0xFFFFu) << 16));
  g1[4] = (int)(tile_d1 & 0xFFFFu);                     // tile_dim1 (tile_dim2=0)
  g1[5] = (int)(stride0 & 0xFFFFFFFFull);               // tensor_dim0_stride[31:0]
  g1[6] = (int)((stride0 >> 32) & 0xFFFFull);           // stride hi (dim1_stride=0)
  g1[7] = 0;
  v4i z4 = {0, 0, 0, 0};
#if defined(__clang_major__) && __clang_major__ >= 23
  v8i z8 = {0, 0, 0, 0, 0, 0, 0, 0};
  __builtin_amdgcn_tensor_load_to_lds(g0, g1, z4, z4, z8, 0);
#else
  __builtin_amdgcn_tensor_load_to_lds(g0, g1, z4, z4, 0);
#endif
}

// ---------------------------------------------------------------------------
// Block-tiled C = A * B^T + bias GEMM with TDM double-buffered LDS staging.
// A:[M,lda] f16, Bm:[N,ldb] f16 (row n = column n of B), C:[M,ldc] f32.
// Block = 128x128 C tile, 256 threads = 8 waves in a 2(M) x 4(N) grid;
// each wave owns 64x32 (4x2 WMMA tiles).  K consumed in 32-wide slabs:
// wave 0 TDM-loads the next A/B 128x32 slabs while all waves compute the
// current one from LDS.  ~64 flop/byte vs ~8 for the naive version.
// Requires M%128==0, N%128==0, K%32==0.
// ---------------------------------------------------------------------------
__global__ void wmma_gemm_tdm(const _Float16* __restrict__ A,
                              const _Float16* __restrict__ Bm,
                              const float* __restrict__ bias,
                              float* __restrict__ C,
                              int M, int N, int K, int lda, int ldb, int ldc)
{
  __shared__ _Float16 ldsA[2][128 * 32];   // 8 KB per buffer
  __shared__ _Float16 ldsB[2][128 * 32];
  const int lane = threadIdx.x & 31;
  const int wave = threadIdx.x >> 5;       // 0..7
  const int wm   = wave >> 2;              // 0..1  (M direction, 64 rows each)
  const int wn   = wave & 3;               // 0..3  (N direction, 32 cols each)
  const int nbk  = N >> 7;
  const int mb   = blockIdx.x / nbk;
  const int nb   = blockIdx.x % nbk;
  const int nlane  = lane & 15;
  const int khalfA = (lane >> 4) << 3;     // 0 or 8
  const int kbB    = (lane >> 4) << 4;     // 0 or 16
  const int brow   = (lane >> 4) << 3;     // C-tile row base (0 or 8)
  const int nk     = K >> 5;

  const unsigned ldsA_addr = (unsigned)(unsigned long long)(const void*)&ldsA[0][0];
  const unsigned ldsB_addr = (unsigned)(unsigned long long)(const void*)&ldsB[0][0];
  const _Float16* Abase = A  + (long)mb * 128 * lda;
  const _Float16* Bbase = Bm + (long)nb * 128 * ldb;

  v8f acc[4][2] = {};

  if (wave == 0) {   // prologue: DMA slab 0
    tdm_load_2d_f16(ldsA_addr, Abase, (unsigned)lda, (unsigned)M, 32, 128,
                    (unsigned long long)lda);
    tdm_load_2d_f16(ldsB_addr, Bbase, (unsigned)ldb, (unsigned)N, 32, 128,
                    (unsigned long long)ldb);
  }

  for (int i = 0; i < nk; ++i) {
    if (wave == 0) {
      if (i + 1 < nk) {     // kick off next slab into the other buffer
        const unsigned buf = (unsigned)((i + 1) & 1) * (128 * 32 * 2);
        tdm_load_2d_f16(ldsA_addr + buf, Abase + (i + 1) * 32, (unsigned)lda,
                        (unsigned)M, 32, 128, (unsigned long long)lda);
        tdm_load_2d_f16(ldsB_addr + buf, Bbase + (i + 1) * 32, (unsigned)ldb,
                        (unsigned)N, 32, 128, (unsigned long long)ldb);
        __builtin_amdgcn_s_wait_tensorcnt(2);   // slab i landed (TDM in-order)
      } else {
        __builtin_amdgcn_s_wait_tensorcnt(0);
      }
    }
    __syncthreads();
    const _Float16* Ab = ldsA[i & 1];
    const _Float16* Bb = ldsB[i & 1];
    v16h bfrag[2];
#pragma unroll
    for (int sn = 0; sn < 2; ++sn)
      bfrag[sn] = load_b_frag(Bb + (wn * 32 + sn * 16 + nlane) * 32, kbB);
#pragma unroll
    for (int sm = 0; sm < 4; ++sm) {
      v16h afrag = load_a_frag(Ab + (wm * 64 + sm * 16 + nlane) * 32, khalfA);
#pragma unroll
      for (int sn = 0; sn < 2; ++sn)
        acc[sm][sn] = __builtin_amdgcn_wmma_f32_16x16x32_f16(
            false, afrag, false, bfrag[sn], (short)0, acc[sm][sn], false, false);
    }
    __syncthreads();   // buffer (i&1) free for reuse at slab i+2
  }

#pragma unroll
  for (int sn = 0; sn < 2; ++sn) {
    const int ncol = nb * 128 + wn * 32 + sn * 16 + nlane;
    const float bv = bias ? bias[ncol] : 0.0f;
#pragma unroll
    for (int sm = 0; sm < 4; ++sm) {
      const int mrow0 = mb * 128 + wm * 64 + sm * 16 + brow;
#pragma unroll
      for (int r = 0; r < 8; ++r)
        C[(long)(mrow0 + r) * ldc + ncol] = acc[sm][sn][r] + bv;
    }
  }
}

// ---------------------------------------------------------------------------
// Embedding gather + LayerNorm -> fp16 rows padded E=300 -> 320.
// One wave per token; job B*T is the SOS row, broadcast to dec_in_h[:,0].
// ---------------------------------------------------------------------------
__global__ void embed_ln(const int* __restrict__ x, const int* __restrict__ sos,
                         const float* __restrict__ emb,
                         const float* __restrict__ g, const float* __restrict__ bta,
                         _Float16* __restrict__ e_h, _Float16* __restrict__ dec_in_h,
                         int B, int T)
{
  const int lane = threadIdx.x & 31;
  const int gw   = (int)((blockIdx.x * blockDim.x + threadIdx.x) >> 5);
  const int ntok = B * T;
  if (gw > ntok) return;
  const int idx = (gw == ntok) ? sos[0] : x[gw];
  const float* row = emb + (long)idx * 300;
  float s = 0.f;
  for (int c = lane; c < 300; c += 32) s += row[c];
  const float mu = wave_reduce(s) * (1.0f / 300.0f);
  float v = 0.f;
  for (int c = lane; c < 300; c += 32) { float d = row[c] - mu; v += d * d; }
  const float rstd = rsqrtf(wave_reduce(v) * (1.0f / 300.0f) + 1e-5f);
  for (int c = lane; c < 320; c += 32) {
    float y = (c < 300) ? (row[c] - mu) * rstd * g[c] + bta[c] : 0.0f;
    _Float16 hy = (_Float16)y;
    if (gw < ntok) {
      e_h[(long)gw * 320 + c] = hy;
      int b = gw / T, t = gw % T;
      if (t < T - 1) dec_in_h[(long)(b * T + t + 1) * 320 + c] = hy;   // shift-right
    } else {
      for (int b = 0; b < B; ++b) dec_in_h[(long)(b * T) * 320 + c] = hy;  // SOS row
    }
  }
}

// ---------------------------------------------------------------------------
// Persistent single-workgroup GRU scan.  32 waves; wave w owns h columns
// [32w, 32w+32).  Per step each wave does 6 WMMA tiles per K-slab
// (r/z/n gates x 2 subtiles) against h staged in LDS as fp16.
// xg:[B*T,3H] precomputed input gates (row m = b*T + t).
// ---------------------------------------------------------------------------
__global__ void gru_scan(const float* __restrict__ xg,
                         const _Float16* __restrict__ Whh,   // [3H,1024] f16
                         const float* __restrict__ bhh,      // [3H]
                         const float* __restrict__ h0,       // [16,1024] or null -> 0
                         float* __restrict__ hn_out,         // [16,1024] or null
                         _Float16* __restrict__ ys_h,        // [B*T,1024] f16 or null
                         int T)
{
  extern __shared__ char smem[];
  _Float16* h_h = (_Float16*)smem;                              // [16][1024] f16
  float*    h_f = (float*)(smem + 16 * 1024 * sizeof(_Float16));// [16][1024] f32
  const int lane = threadIdx.x & 31;
  const int wave = threadIdx.x >> 5;                            // 0..31
  for (int i = threadIdx.x; i < 16 * 1024; i += blockDim.x) {
    float v = h0 ? h0[i] : 0.0f;
    h_f[i] = v;
    h_h[i] = (_Float16)v;
  }
  __syncthreads();
  const int nlane   = lane & 15;
  const int khalfA  = (lane >> 4) << 3;
  const int kbB     = (lane >> 4) << 4;
  const int colbase = wave * 32;
  const int brow    = (lane >> 4) << 3;            // C-tile row base (0 or 8)
  const _Float16* hrow = h_h + (long)nlane * 1024; // A row for this lane (M = batch nlane)

  for (int t = 0; t < T; ++t) {
    v8f c[6] = {};
    for (int kk = 0; kk < 1024; kk += 32) {
      v16h a = load_a_frag(hrow + kk, khalfA);
#pragma unroll
      for (int gte = 0; gte < 3; ++gte) {
#pragma unroll
        for (int s = 0; s < 2; ++s) {
          const _Float16* wr = Whh + (long)(gte * 1024 + colbase + s * 16 + nlane) * 1024;
          c[gte * 2 + s] = __builtin_amdgcn_wmma_f32_16x16x32_f16(
              false, a, false, load_b_frag(wr + kk, kbB), (short)0, c[gte * 2 + s], false, false);
        }
      }
    }
    float hnew[2][8];
#pragma unroll
    for (int s = 0; s < 2; ++s) {
      const int hcol = colbase + s * 16 + nlane;
      const float br = bhh[hcol], bz = bhh[1024 + hcol], bn = bhh[2048 + hcol];
#pragma unroll
      for (int r = 0; r < 8; ++r) {
        const int b = brow + r;
        const float* xrow = xg + (long)(b * T + t) * 3072;
        float gr = c[s][r]     + br + xrow[hcol];
        float gz = c[2 + s][r] + bz + xrow[1024 + hcol];
        float gh = c[4 + s][r] + bn;                 // recurrent n-gate (incl. bhh_n)
        float gx = xrow[2048 + hcol];
        float rg = 1.0f / (1.0f + __expf(-gr));
        float zg = 1.0f / (1.0f + __expf(-gz));
        float ng = tanhf(gx + rg * gh);
        float hv = (1.0f - zg) * ng + zg * h_f[b * 1024 + hcol];
        hnew[s][r] = hv;
        if (ys_h) ys_h[(long)(b * T + t) * 1024 + hcol] = (_Float16)hv;
      }
    }
    __syncthreads();   // all reads of h done before anyone rewrites it
#pragma unroll
    for (int s = 0; s < 2; ++s) {
      const int hcol = colbase + s * 16 + nlane;
#pragma unroll
      for (int r = 0; r < 8; ++r) {
        const int b = brow + r;
        h_f[b * 1024 + hcol] = hnew[s][r];
        h_h[b * 1024 + hcol] = (_Float16)hnew[s][r];
      }
    }
    __syncthreads();
  }
  if (hn_out)
    for (int i = threadIdx.x; i < 16 * 1024; i += blockDim.x) hn_out[i] = h_f[i];
}

// ---------------------------------------------------------------------------
// VAE heads: per (b, j) wave computes mu/logvar dots over H=1024, reparam sample,
// writes mu/logvar/style/content to d_out slices and [style|content] to zcat.
// ---------------------------------------------------------------------------
__global__ void vae_heads(const float* __restrict__ hn,
                          const float* __restrict__ eps_s, const float* __restrict__ eps_c,
                          const float* __restrict__ mus_W, const float* __restrict__ mus_b,
                          const float* __restrict__ vars_W, const float* __restrict__ vars_b,
                          const float* __restrict__ muc_W, const float* __restrict__ muc_b,
                          const float* __restrict__ varc_W, const float* __restrict__ varc_b,
                          float* __restrict__ o_style, float* __restrict__ o_content,
                          float* __restrict__ o_mus, float* __restrict__ o_lvs,
                          float* __restrict__ o_muc, float* __restrict__ o_lvc,
                          float* __restrict__ zcat)
{
  const int lane = threadIdx.x & 31;
  const int gw   = (int)((blockIdx.x * blockDim.x + threadIdx.x) >> 5);
  if (gw >= 16 * 256) return;
  const int b = gw >> 8, j = gw & 255;
  const float *Wm, *Wv; float bm, bv2, ep; float *omu, *olv, *osmp;
  if (j < 64) {
    Wm = mus_W + (long)j * 1024;  Wv = vars_W + (long)j * 1024;
    bm = mus_b[j]; bv2 = vars_b[j]; ep = eps_s[b * 64 + j];
    omu = o_mus + b * 64 + j; olv = o_lvs + b * 64 + j; osmp = o_style + b * 64 + j;
  } else {
    const int jj = j - 64;
    Wm = muc_W + (long)jj * 1024; Wv = varc_W + (long)jj * 1024;
    bm = muc_b[jj]; bv2 = varc_b[jj]; ep = eps_c[b * 192 + jj];
    omu = o_muc + b * 192 + jj; olv = o_lvc + b * 192 + jj; osmp = o_content + b * 192 + jj;
  }
  float am = 0.f, av = 0.f;
  for (int k = lane; k < 1024; k += 32) {
    float h = hn[b * 1024 + k];
    am += Wm[k] * h;
    av += Wv[k] * h;
  }
  am = wave_reduce(am); av = wave_reduce(av);
  if (lane == 0) {
    float mu = am + bm, lv = av + bv2;
    float smp = mu + ep * __expf(0.5f * lv);
    *omu = mu; *olv = lv; *osmp = smp;
    zcat[b * 256 + j] = smp;
  }
}

__global__ void fc_z(const float* __restrict__ zcat, const float* __restrict__ fc_W,
                     const float* __restrict__ fc_b, float* __restrict__ z)
{
  const int lane = threadIdx.x & 31;
  const int gw   = (int)((blockIdx.x * blockDim.x + threadIdx.x) >> 5);
  if (gw >= 16 * 1024) return;
  const int b = gw >> 10, n = gw & 1023;
  float acc = 0.f;
  for (int k = lane; k < 256; k += 32) acc += zcat[b * 256 + k] * fc_W[(long)n * 256 + k];
  acc = wave_reduce(acc);
  if (lane == 0) z[b * 1024 + n] = acc + fc_b[n];
}

__global__ void cvt_f32_f16_pad(const float* __restrict__ src, _Float16* __restrict__ dst,
                                long rows, int scols, int dcols)
{
  long i = (long)blockIdx.x * blockDim.x + threadIdx.x;
  if (i >= rows * (long)dcols) return;
  long r = i / dcols; int cc = (int)(i % dcols);
  dst[i] = (cc < scols) ? (_Float16)src[r * (long)scols + cc] : (_Float16)0.0f;
}

// ---------------------------------------------------------------------------
extern "C" void kernel_launch(void* const* d_in, const int* in_sizes, int n_in,
                              void* d_out, int out_size, void* d_ws, size_t ws_size,
                              hipStream_t stream)
{
  (void)in_sizes; (void)n_in; (void)out_size; (void)ws_size;
  constexpr int B = 16, T = 128, V = 32000, E = 300, Ep = 320, H = 1024, S = 64, Cc = 192;
  constexpr int M  = B * T;     // 2048
  constexpr int G3 = 3 * H;     // 3072

  const int*   x       = (const int*)  d_in[0];
  const int*   sos     = (const int*)  d_in[1];
  const float* eps_s   = (const float*)d_in[2];
  const float* eps_c   = (const float*)d_in[3];
  const float* emb     = (const float*)d_in[4];
  const float* ln_g    = (const float*)d_in[5];
  const float* ln_bp   = (const float*)d_in[6];
  const float* encWih  = (const float*)d_in[7];
  const float* encWhh  = (const float*)d_in[8];
  const float* enc_bih = (const float*)d_in[9];
  const float* enc_bhh = (const float*)d_in[10];
  const float* mus_W   = (const float*)d_in[11];
  const float* mus_b   = (const float*)d_in[12];
  const float* vars_W  = (const float*)d_in[13];
  const float* vars_b  = (const float*)d_in[14];
  const float* muc_W   = (const float*)d_in[15];
  const float* muc_b   = (const float*)d_in[16];
  const float* varc_W  = (const float*)d_in[17];
  const float* varc_b  = (const float*)d_in[18];
  const float* fc_W    = (const float*)d_in[19];
  const float* fc_b    = (const float*)d_in[20];
  const float* decWih  = (const float*)d_in[21];
  const float* decWhh  = (const float*)d_in[22];
  const float* dec_bih = (const float*)d_in[23];
  const float* dec_bhh = (const float*)d_in[24];
  const float* out_W   = (const float*)d_in[25];
  const float* out_b   = (const float*)d_in[26];

  char* p = (char*)d_ws;
  auto carve = [&](size_t bytes) { char* r = p; p += (bytes + 255) & ~(size_t)255; return (void*)r; };
  _Float16* e_h      = (_Float16*)carve((size_t)M  * Ep * 2);
  _Float16* dec_in_h = (_Float16*)carve((size_t)M  * Ep * 2);
  _Float16* encWih_h = (_Float16*)carve((size_t)G3 * Ep * 2);
  _Float16* decWih_h = (_Float16*)carve((size_t)G3 * Ep * 2);
  _Float16* encWhh_h = (_Float16*)carve((size_t)G3 * H  * 2);
  _Float16* decWhh_h = (_Float16*)carve((size_t)G3 * H  * 2);
  _Float16* outW_h   = (_Float16*)carve((size_t)V  * H  * 2);
  float*    xg_enc   = (float*)carve((size_t)M * G3 * 4);
  float*    xg_dec   = (float*)carve((size_t)M * G3 * 4);
  float*    hn       = (float*)carve((size_t)B * H * 4);
  float*    zcat     = (float*)carve((size_t)B * (S + Cc) * 4);
  float*    zvec     = (float*)carve((size_t)B * H * 4);
  _Float16* ys_h     = (_Float16*)carve((size_t)M * H * 2);

  float* logits    = (float*)d_out;
  float* o_style   = logits + (size_t)M * V;
  float* o_content = o_style   + B * S;
  float* o_mus     = o_content + B * Cc;
  float* o_lvs     = o_mus     + B * S;
  float* o_muc     = o_lvs     + B * S;
  float* o_lvc     = o_muc     + B * Cc;

  auto cdiv = [](long a, long b) { return (unsigned)((a + b - 1) / b); };

  // 1) weights -> fp16 (E padded to 320 so K is a multiple of 32)
  cvt_f32_f16_pad<<<cdiv((long)G3 * Ep, 256), 256, 0, stream>>>(encWih, encWih_h, G3, E, Ep);
  cvt_f32_f16_pad<<<cdiv((long)G3 * Ep, 256), 256, 0, stream>>>(decWih, decWih_h, G3, E, Ep);
  cvt_f32_f16_pad<<<cdiv((long)G3 * H,  256), 256, 0, stream>>>(encWhh, encWhh_h, G3, H, H);
  cvt_f32_f16_pad<<<cdiv((long)G3 * H,  256), 256, 0, stream>>>(decWhh, decWhh_h, G3, H, H);
  cvt_f32_f16_pad<<<cdiv((long)V  * H,  256), 256, 0, stream>>>(out_W,  outW_h,   V,  H, H);

  // 2) embedding + LayerNorm -> e_h, shifted dec_in_h (fp16)
  embed_ln<<<cdiv((long)(M + 1) * 32, 256), 256, 0, stream>>>(x, sos, emb, ln_g, ln_bp,
                                                              e_h, dec_in_h, B, T);

  // 3) input-gate GEMMs: xg = e @ Wih^T + bih   [2048 x 3072, K=320]
  wmma_gemm_tdm<<<(M / 128) * (G3 / 128), 256, 0, stream>>>(e_h,      encWih_h, enc_bih,
                                                            xg_enc, M, G3, Ep, Ep, Ep, G3);
  wmma_gemm_tdm<<<(M / 128) * (G3 / 128), 256, 0, stream>>>(dec_in_h, decWih_h, dec_bih,
                                                            xg_dec, M, G3, Ep, Ep, Ep, G3);

  // 4) encoder GRU scan (h0 = 0) -> hn
  gru_scan<<<1, 1024, 96 * 1024, stream>>>(xg_enc, encWhh_h, enc_bhh, nullptr, hn, nullptr, T);

  // 5) VAE heads + reparameterization + fc -> zvec (decoder h0)
  vae_heads<<<cdiv((long)B * (S + Cc) * 32, 256), 256, 0, stream>>>(
      hn, eps_s, eps_c, mus_W, mus_b, vars_W, vars_b, muc_W, muc_b, varc_W, varc_b,
      o_style, o_content, o_mus, o_lvs, o_muc, o_lvc, zcat);
  fc_z<<<cdiv((long)B * H * 32, 256), 256, 0, stream>>>(zcat, fc_W, fc_b, zvec);

  // 6) decoder GRU scan (h0 = z) -> ys_h (fp16)
  gru_scan<<<1, 1024, 96 * 1024, stream>>>(xg_dec, decWhh_h, dec_bhh, zvec, nullptr, ys_h, T);

  // 7) logits = ys @ out_W^T + out_b   [2048 x 32000, K=1024]  (dominant GEMM)
  wmma_gemm_tdm<<<(M / 128) * (V / 128), 256, 0, stream>>>(ys_h, outW_h, out_b,
                                                           logits, M, V, H, H, H, V);
}
